// GFCN_12524124635537
// MI455X (gfx1250) — matrix operand order repaired and checked by reading, not compiled, not run on hardware
//
#include <hip/hip_runtime.h>
#include <hip/hip_bf16.h>
#include <math.h>
#include <stdint.h>

// Problem sizes (fixed by the reference)
#define N1n 98304
#define N2n 49152
#define N3n 24576
#define E1n 393216
#define E2n 196608
#define E3n 98304

typedef __attribute__((ext_vector_type(16))) _Float16 v16h;
typedef __attribute__((ext_vector_type(8)))  float    v8f;

// ---------------------------------------------------------------------------
// Spline basis: K=5, 2D, degree-1 -> 4 corners, kernel grid 5x5 (=25).
// BITS = [[0,0],[1,0],[0,1],[1,1]], STRIDE = [1,5]
// ---------------------------------------------------------------------------
__device__ __forceinline__ void spline_basis4(float p0, float p1,
                                              float basis[4], int kidx[4]) {
  float v0 = p0 * 4.0f, v1 = p1 * 4.0f;
  float k0 = fminf(fmaxf(floorf(v0), 0.0f), 3.0f);
  float k1 = fminf(fmaxf(floorf(v1), 0.0f), 3.0f);
  float f0 = v0 - k0, f1 = v1 - k1;
  int i0 = (int)k0, i1 = (int)k1;
#pragma unroll
  for (int s = 0; s < 4; ++s) {
    int b0 = s & 1, b1 = (s >> 1) & 1;
    basis[s] = (b0 ? f0 : 1.0f - f0) * (b1 ? f1 : 1.0f - f1);
    kidx[s]  = (i0 + b0) + (i1 + b1) * 5;
  }
}

__device__ __forceinline__ void atomicMaxFloat(float* addr, float val) {
  if (!(val < 0.0f)) {               // val >= 0 or NaN
    atomicMax((int*)addr, __float_as_int(val));
  } else {
    atomicMin((unsigned int*)addr, __float_as_uint(val));
  }
}

// ---------------------------------------------------------------------------
// One-time weight prep: f32 (25 x IN x OUT) -> f16, pre-swizzled into the
// B-fragment LDS order (contiguous k = 16*half + t), one contiguous chunk per 16-wide output tile:
//   Wswz[nt][ ((k*CH+q)*32 + blane)*16 + t ],
//   blane = col + 16*((krow%32)>=16), t = krow%16, q = krow/32
// so staging into LDS becomes a raw byte copy (async-to-LDS friendly).
// ---------------------------------------------------------------------------
template<int IN, int OUT>
__global__ void prep_weights(const float* __restrict__ W, _Float16* __restrict__ Wswz)
{
  constexpr int NK = 25, CH = IN / 32;
  int i = blockIdx.x * blockDim.x + threadIdx.x;      // over NT*NK*IN*16
  if (i >= (OUT / 16) * NK * IN * 16) return;
  int nt   = i / (NK * IN * 16);
  int rem  = i % (NK * IN * 16);
  int col  = rem & 15;
  int krow = (rem >> 4) % IN;
  int kker = rem / (IN * 16);
  float wv = W[((long)kker * IN + krow) * OUT + nt * 16 + col];
  int q = krow >> 5, kin = krow & 31;
  int blane = col + 16 * (kin >> 4);
  int t = kin & 15;
  Wswz[(size_t)nt * (NK * IN * 16) + (((kker * CH + q) * 32) + blane) * 16 + t] =
      (_Float16)wv;
}

// ---------------------------------------------------------------------------
// WMMA spline-conv edge kernel.  128 threads = 4 waves; each wave owns a tile
// of 16 edges.  PRECONDITION: E % 64 == 0 (all tiles full) -- true for all
// three edge sets here; the scatter is branch-free straight-line atomics.
// Per 16-wide output tile:
//   async-copy the pre-swizzled W chunk into LDS (global_load_async_to_lds),
//   for k in 0..24 (unroll 2: next iteration's independent WMMA fills the
//   WMMA->VALU hazard window without blowing registers):
//     C_k = A_edges (16xIN f16) x W_k (INx16 f16) via v_wmma_f32_16x16x32_f16
//     (chained over IN/32 K-chunks), then select-accumulate with per-edge
//     basis weights; finally scatter via f32 atomics.
// ---------------------------------------------------------------------------
template<int IN, int OUT>
__global__ __launch_bounds__(128, 4)
void spline_wmma_kernel(const float*    __restrict__ X,       // N x IN
                        const int*      __restrict__ src,     // E
                        const int*      __restrict__ dst,     // E
                        const float*    __restrict__ pseudo,  // E x 2
                        const _Float16* __restrict__ Wswz,    // NT x (25*IN*16)
                        float*          __restrict__ agg,     // N x OUT (accum)
                        int E)
{
  constexpr int NK = 25;
  constexpr int CH = IN / 32;             // K chunks of 32
  constexpr int NT = OUT / 16;            // output tiles of 16
  constexpr int CHUNKB = NK * IN * 16 * 2;  // staged bytes per output tile

  __shared__ __align__(32) _Float16 wlds[NK * IN * 16]; // B fragments
  __shared__ float wsel[4][16 * NK];                    // per-edge kernel weights
  __shared__ int   ssrc[4][16];
  __shared__ int   sdst[4][16];

  const int tid  = threadIdx.x;
  const int wave = tid >> 5;
  const int lane = tid & 31;
  const int l16  = lane & 15;
  const int hi   = lane >> 4;

  const int ebase = (blockIdx.x * 4 + wave) * 16;
  (void)E;  // E % 64 == 0 guaranteed; every tile is full

  // zero this wave's selection table
  for (int i = lane; i < 16 * NK; i += 32) wsel[wave][i] = 0.0f;

  // lanes 0..15 of each wave: per-edge basis -> selection table
  if (hi == 0) {
    int e = ebase + l16;
    ssrc[wave][l16] = src[e];
    sdst[wave][l16] = dst[e];
    float basis[4]; int kidx[4];
    spline_basis4(pseudo[2 * e + 0], pseudo[2 * e + 1], basis, kidx);
#pragma unroll
    for (int s = 0; s < 4; ++s)
      wsel[wave][l16 * NK + kidx[s]] += basis[s];
  }
  __syncthreads();

  // Build A fragments (16 edges x IN, f16), contiguous k order (k = 16*hi + s), same as B:
  //   lane<16 : halfs 0..7 -> K q*32+0..7,   halfs 8..15 -> K q*32+8..15
  //   lane>=16: halfs 0..7 -> K q*32+16..23, halfs 8..15 -> K q*32+24..31
  // Each 8-K group is contiguous in memory -> 4 aligned float4 loads/chunk.
  v16h areg[CH];
  {
    const float* xr = X + (long)ssrc[wave][l16] * IN;
#pragma unroll
    for (int q = 0; q < CH; ++q) {
      const float4* p0 = (const float4*)(xr + q * 32 + 16 * hi);       // K base+0..7
      const float4* p1 = (const float4*)(xr + q * 32 + 16 * hi + 8);   // K base+8..15
      float4 u0 = p0[0], u1 = p0[1];
      float4 v0 = p1[0], v1 = p1[1];
      v16h a;
      a[0]  = (_Float16)u0.x; a[1]  = (_Float16)u0.y;
      a[2]  = (_Float16)u0.z; a[3]  = (_Float16)u0.w;
      a[4]  = (_Float16)u1.x; a[5]  = (_Float16)u1.y;
      a[6]  = (_Float16)u1.z; a[7]  = (_Float16)u1.w;
      a[8]  = (_Float16)v0.x; a[9]  = (_Float16)v0.y;
      a[10] = (_Float16)v0.z; a[11] = (_Float16)v0.w;
      a[12] = (_Float16)v1.x; a[13] = (_Float16)v1.y;
      a[14] = (_Float16)v1.z; a[15] = (_Float16)v1.w;
      areg[q] = a;
    }
  }

  for (int nt = 0; nt < NT; ++nt) {
    __syncthreads();   // all waves done reading previous tile's wlds

    // Async DMA the pre-swizzled chunk into LDS: memory -> LDS, no VGPRs.
    {
      const char* gsrc = (const char*)Wswz + (size_t)nt * CHUNKB;
      unsigned lbase = (unsigned)(uintptr_t)&wlds[0];  // low 32b of flat = LDS offset
      for (int i = tid * 16; i < CHUNKB; i += 128 * 16) {
        unsigned lofs = lbase + (unsigned)i;
        unsigned long long ga = (unsigned long long)(uintptr_t)(gsrc + i);
        asm volatile("global_load_async_to_lds_b128 %0, %1, off"
                     :: "v"(lofs), "v"(ga) : "memory");
      }
      asm volatile("s_wait_asynccnt 0x0" ::: "memory");
    }
    __syncthreads();

    float macc[8];
#pragma unroll
    for (int r = 0; r < 8; ++r) macc[r] = 0.0f;

    // unroll 2: one extra B fragment + C tile in flight (~24 VGPRs) lets the
    // independent next WMMA fill the WMMA->VALU hazard slots; full unroll
    // would spill C tiles under the 256-VGPR cap.
#pragma unroll 2
    for (int k = 0; k < NK; ++k) {
      v8f c = {};
#pragma unroll
      for (int q = 0; q < CH; ++q) {
        const v16h* bp = (const v16h*)&wlds[(((k * CH + q) * 32) + lane) * 16];
        c = __builtin_amdgcn_wmma_f32_16x16x32_f16(
            false, areg[q], false, *bp, (short)0, c, false, false);
      }
      // C layout: reg r of lane -> edge row (r + 8*hi), column l16
#pragma unroll
      for (int r = 0; r < 8; ++r) {
        int m = r + 8 * hi;
        macc[r] += wsel[wave][m * NK + k] * c[r];
      }
    }

    // Branch-free scatter: all tiles are full (E % 64 == 0).
#pragma unroll
    for (int r = 0; r < 8; ++r) {
      int m = r + 8 * hi;
      int d = sdst[wave][m];
      atomicAdd(&agg[(long)d * OUT + nt * 16 + l16], macc[r]);
    }
  }
}

// ---------------------------------------------------------------------------
// Degenerate-shape spline conv layers (IN=1 or OUT=1): scalar edge kernels.
// ---------------------------------------------------------------------------
__global__ void layer1_edge(const float* __restrict__ x, const int* __restrict__ ei,
                            const float* __restrict__ pseudo,
                            const float* __restrict__ W,   // 25 x 1 x 32
                            float* __restrict__ agg, int E)
{
  __shared__ float wl[25 * 32];
  for (int i = threadIdx.x; i < 25 * 32; i += blockDim.x) wl[i] = W[i];
  __syncthreads();
  int e = blockIdx.x * blockDim.x + threadIdx.x;
  if (e >= E) return;
  int s = ei[e], d = ei[E + e];
  float basis[4]; int kidx[4];
  spline_basis4(pseudo[2 * e], pseudo[2 * e + 1], basis, kidx);
  float xs = x[s];
#pragma unroll
  for (int o = 0; o < 32; ++o) {
    float m = 0.0f;
#pragma unroll
    for (int c = 0; c < 4; ++c) m += basis[c] * wl[kidx[c] * 32 + o];
    atomicAdd(&agg[(long)d * 32 + o], xs * m);
  }
}

__global__ void layer4_edge(const float* __restrict__ x, const int* __restrict__ ei,
                            const float* __restrict__ pseudo,
                            const float* __restrict__ W,   // 25 x 32 x 1
                            float* __restrict__ agg, int E)
{
  __shared__ float wl[25 * 32];
  for (int i = threadIdx.x; i < 25 * 32; i += blockDim.x) wl[i] = W[i];
  __syncthreads();
  int e = blockIdx.x * blockDim.x + threadIdx.x;
  if (e >= E) return;
  int s = ei[e], d = ei[E + e];
  float basis[4]; int kidx[4];
  spline_basis4(pseudo[2 * e], pseudo[2 * e + 1], basis, kidx);
  const float* xr = x + (long)s * 32;
  float acc = 0.0f;
#pragma unroll
  for (int c = 0; c < 4; ++c) {
    const float* wk = &wl[kidx[c] * 32];
    float dotv = 0.0f;
#pragma unroll
    for (int i = 0; i < 32; ++i) dotv += xr[i] * wk[i];
    acc += basis[c] * dotv;
  }
  atomicAdd(&agg[d], acc);
}

// ---------------------------------------------------------------------------
// Elementwise / segment helper kernels
// ---------------------------------------------------------------------------
template<int IN, int OUT>
__global__ void finalize_kernel(const float* __restrict__ agg, const float* __restrict__ deg,
                                const float* __restrict__ X, const float* __restrict__ root,
                                const float* __restrict__ bias, float* __restrict__ h, int N)
{
  int i = blockIdx.x * blockDim.x + threadIdx.x;
  if (i >= N * OUT) return;
  int n = i / OUT, o = i % OUT;
  float acc = agg[i] / fmaxf(deg[n], 1.0f);
  const float* xr = X + (long)n * IN;
  float r = 0.0f;
#pragma unroll 4
  for (int k = 0; k < IN; ++k) r += xr[k] * root[k * OUT + o];
  float v = acc + r + bias[o];
  h[i] = (v > 0.0f) ? v : expm1f(v);           // ELU
}

__global__ void count_deg(const int* __restrict__ idx, float* __restrict__ deg, int n)
{
  int i = blockIdx.x * blockDim.x + threadIdx.x;
  if (i < n) atomicAdd(&deg[idx[i]], 1.0f);
}

__global__ void scatter_add(const float* __restrict__ x, const int* __restrict__ cl,
                            float* __restrict__ g, int N, int F)
{
  int i = blockIdx.x * blockDim.x + threadIdx.x;
  if (i >= N * F) return;
  int n = i / F, f = i % F;
  atomicAdd(&g[(long)cl[n] * F + f], x[i]);
}

__global__ void pweights_div(const float* __restrict__ x, const int* __restrict__ cl,
                             const float* __restrict__ g, float* __restrict__ w, int N, int F)
{
  int i = blockIdx.x * blockDim.x + threadIdx.x;
  if (i >= N * F) return;
  int n = i / F, f = i % F;
  float v = x[i] / g[(long)cl[n] * F + f];
  w[i] = (v != v) ? 0.0f : v;
}

__global__ void fill_kernel(float* __restrict__ p, float v, int n)
{
  int i = blockIdx.x * blockDim.x + threadIdx.x;
  if (i < n) p[i] = v;
}

__global__ void scatter_max(const float* __restrict__ h, const int* __restrict__ cl,
                            float* __restrict__ m, int N, int F)
{
  int i = blockIdx.x * blockDim.x + threadIdx.x;
  if (i >= N * F) return;
  int n = i / F, f = i % F;
  atomicMaxFloat(&m[(long)cl[n] * F + f], h[i]);
}

__global__ void fix_max(float* __restrict__ m, int n)
{
  int i = blockIdx.x * blockDim.x + threadIdx.x;
  if (i < n) { float v = m[i]; if (!__builtin_isfinite(v)) m[i] = 0.0f; }
}

__global__ void div_rows(float* __restrict__ s, const float* __restrict__ c, int N, int F)
{
  int i = blockIdx.x * blockDim.x + threadIdx.x;
  if (i >= N * F) return;
  s[i] = s[i] / fmaxf(c[i / F], 1.0f);
}

__global__ void cartesian_kernel(const float* __restrict__ pos, const int* __restrict__ ei,
                                 float* __restrict__ out, int E)
{
  int e = blockIdx.x * blockDim.x + threadIdx.x;
  if (e >= E) return;
  int row = ei[e], col = ei[E + e];
  float c0 = pos[2 * row + 0] - pos[2 * col + 0];
  float c1 = pos[2 * row + 1] - pos[2 * col + 1];
  float m = fmaxf(fabsf(c0), fabsf(c1));
  float inv = (m > 0.0f) ? (1.0f / (2.0f * m)) : 0.0f;
  out[2 * e + 0] = c0 * inv + 0.5f;
  out[2 * e + 1] = c1 * inv + 0.5f;
}

__global__ void gather_mul(const float* __restrict__ h, const int* __restrict__ cl,
                           const float* __restrict__ w, float* __restrict__ out, int N, int F)
{
  int i = blockIdx.x * blockDim.x + threadIdx.x;
  if (i >= N * F) return;
  int n = i / F, f = i % F;
  out[i] = h[(long)cl[n] * F + f] * w[i];
}

// ---------------------------------------------------------------------------
// Host orchestration (graph-capture safe: kernels + hipMemsetAsync only)
// ---------------------------------------------------------------------------
extern "C" void kernel_launch(void* const* d_in, const int* in_sizes, int n_in,
                              void* d_out, int out_size, void* d_ws, size_t ws_size,
                              hipStream_t stream)
{
  (void)in_sizes; (void)n_in; (void)out_size; (void)ws_size;

  const float* x    = (const float*)d_in[0];
  const float* pos1 = (const float*)d_in[1];
  const float* ea1  = (const float*)d_in[2];
  const int*   ei1  = (const int*)  d_in[3];
  const int*   cl1  = (const int*)  d_in[4];
  const int*   ei2  = (const int*)  d_in[5];
  const int*   cl2  = (const int*)  d_in[6];
  const int*   ei3  = (const int*)  d_in[7];
  const float* W1   = (const float*)d_in[8];
  const float* r1   = (const float*)d_in[9];
  const float* b1   = (const float*)d_in[10];
  const float* W2   = (const float*)d_in[11];
  const float* r2   = (const float*)d_in[12];
  const float* b2   = (const float*)d_in[13];
  const float* W3   = (const float*)d_in[14];
  const float* r3   = (const float*)d_in[15];
  const float* b3   = (const float*)d_in[16];
  const float* W4   = (const float*)d_in[17];
  const float* r4   = (const float*)d_in[18];
  const float* b4   = (const float*)d_in[19];
  float* out = (float*)d_out;

  float* base = (float*)d_ws;
  size_t off = 0;
  auto A = [&](size_t nelem) { float* r = base + off; off += nelem; return r; };

  float* h1   = A((size_t)N1n * 32);
  float* deg1 = A(N1n);
  float* w1   = A(N1n);
  float* g1   = A(N2n);
  float* x2   = A((size_t)N2n * 32);
  float* pos2 = A((size_t)N2n * 2);
  float* cnt2 = A(N2n);
  float* ea2  = A((size_t)E2n * 2);
  float* h2   = A((size_t)N2n * 64);
  float* deg2 = A(N2n);
  float* w2   = A((size_t)N2n * 32);
  float* g2   = A((size_t)N3n * 32);
  float* x3   = A((size_t)N3n * 64);
  float* pos3 = A((size_t)N3n * 2);
  float* cnt3 = A(N3n);
  float* ea3  = A((size_t)E3n * 2);
  float* h3   = A((size_t)N3n * 32);
  float* deg3 = A(N3n);
  float* xr2  = A((size_t)N2n * 32);
  float* h4   = A(N2n);
  float* deg4 = A(N2n);
  _Float16* w2swz = (_Float16*)A(25 * 32 * 64 / 2);  // 25*32*64 halfs
  _Float16* w3swz = (_Float16*)A(25 * 64 * 32 / 2);  // 25*64*32 halfs

  const int B = 256;
  auto G = [](int n) { return dim3((unsigned)((n + 255) / 256)); };
#define ZEROF(p, n) hipMemsetAsync((p), 0, (size_t)(n) * sizeof(float), stream)

  // ---- Pre-swizzle W2/W3 into f16 B-fragment chunks (async-copy friendly) ----
  prep_weights<32, 64><<<G(25 * 32 * 64), B, 0, stream>>>(W2, w2swz);
  prep_weights<64, 32><<<G(25 * 64 * 32), B, 0, stream>>>(W3, w3swz);

  // ---- Layer 1: spline_conv(x, ei1, ea1, W1) -> h1 (N1 x 32), ELU ----
  ZEROF(h1, (size_t)N1n * 32); ZEROF(deg1, N1n);
  layer1_edge<<<G(E1n), B, 0, stream>>>(x, ei1, ea1, W1, h1, E1n);
  count_deg<<<G(E1n), B, 0, stream>>>(ei1 + E1n, deg1, E1n);
  finalize_kernel<1, 32><<<G(N1n * 32), B, 0, stream>>>(h1, deg1, x, r1, b1, h1, N1n);

  // ---- pweights w1 = x / seg_sum(x, cl1) ----
  ZEROF(g1, N2n);
  scatter_add<<<G(N1n), B, 0, stream>>>(x, cl1, g1, N1n, 1);
  pweights_div<<<G(N1n), B, 0, stream>>>(x, cl1, g1, w1, N1n, 1);

  // ---- x2 = seg_max0(h1, cl1) ----
  fill_kernel<<<G(N2n * 32), B, 0, stream>>>(x2, -INFINITY, N2n * 32);
  scatter_max<<<G(N1n * 32), B, 0, stream>>>(h1, cl1, x2, N1n, 32);
  fix_max<<<G(N2n * 32), B, 0, stream>>>(x2, N2n * 32);

  // ---- pos2 = seg_mean(pos1, cl1); ea2 = cartesian(pos2, ei2) ----
  ZEROF(pos2, (size_t)N2n * 2); ZEROF(cnt2, N2n);
  scatter_add<<<G(N1n * 2), B, 0, stream>>>(pos1, cl1, pos2, N1n, 2);
  count_deg<<<G(N1n), B, 0, stream>>>(cl1, cnt2, N1n);
  div_rows<<<G(N2n * 2), B, 0, stream>>>(pos2, cnt2, N2n, 2);
  cartesian_kernel<<<G(E2n), B, 0, stream>>>(pos2, ei2, ea2, E2n);

  // ---- Layer 2 (WMMA): spline_conv(x2, ei2, ea2, W2) -> h2 (N2 x 64), ELU ----
  ZEROF(h2, (size_t)N2n * 64); ZEROF(deg2, N2n);
  spline_wmma_kernel<32, 64><<<dim3(E2n / 64), 128, 0, stream>>>(x2, ei2, ei2 + E2n, ea2, w2swz, h2, E2n);
  count_deg<<<G(E2n), B, 0, stream>>>(ei2 + E2n, deg2, E2n);
  finalize_kernel<32, 64><<<G(N2n * 64), B, 0, stream>>>(h2, deg2, x2, r2, b2, h2, N2n);

  // ---- pweights w2 = x2 / seg_sum(x2, cl2) ----
  ZEROF(g2, (size_t)N3n * 32);
  scatter_add<<<G(N2n * 32), B, 0, stream>>>(x2, cl2, g2, N2n, 32);
  pweights_div<<<G(N2n * 32), B, 0, stream>>>(x2, cl2, g2, w2, N2n, 32);

  // ---- x3 = seg_max0(h2, cl2) ----
  fill_kernel<<<G(N3n * 64), B, 0, stream>>>(x3, -INFINITY, N3n * 64);
  scatter_max<<<G(N2n * 64), B, 0, stream>>>(h2, cl2, x3, N2n, 64);
  fix_max<<<G(N3n * 64), B, 0, stream>>>(x3, N3n * 64);

  // ---- pos3 = seg_mean(pos2, cl2); ea3 = cartesian(pos3, ei3) ----
  ZEROF(pos3, (size_t)N3n * 2); ZEROF(cnt3, N3n);
  scatter_add<<<G(N2n * 2), B, 0, stream>>>(pos2, cl2, pos3, N2n, 2);
  count_deg<<<G(N2n), B, 0, stream>>>(cl2, cnt3, N2n);
  div_rows<<<G(N3n * 2), B, 0, stream>>>(pos3, cnt3, N3n, 2);
  cartesian_kernel<<<G(E3n), B, 0, stream>>>(pos3, ei3, ea3, E3n);

  // ---- Layer 3 (WMMA): spline_conv(x3, ei3, ea3, W3) -> h3 (N3 x 32), ELU ----
  ZEROF(h3, (size_t)N3n * 32); ZEROF(deg3, N3n);
  spline_wmma_kernel<64, 32><<<dim3(E3n / 64), 128, 0, stream>>>(x3, ei3, ei3 + E3n, ea3, w3swz, h3, E3n);
  count_deg<<<G(E3n), B, 0, stream>>>(ei3 + E3n, deg3, E3n);
  finalize_kernel<64, 32><<<G(N3n * 32), B, 0, stream>>>(h3, deg3, x3, r3, b3, h3, N3n);

  // ---- xr2 = h3[cl2] * w2 ----
  gather_mul<<<G(N2n * 32), B, 0, stream>>>(h3, cl2, w2, xr2, N2n, 32);

  // ---- Layer 4: spline_conv(xr2, ei2, ea2, W4) -> h4 (N2 x 1), ELU ----
  ZEROF(h4, N2n); ZEROF(deg4, N2n);
  layer4_edge<<<G(E2n), B, 0, stream>>>(xr2, ei2, ea2, W4, h4, E2n);
  count_deg<<<G(E2n), B, 0, stream>>>(ei2 + E2n, deg4, E2n);
  finalize_kernel<32, 1><<<G(N2n), B, 0, stream>>>(h4, deg4, xr2, r4, b4, h4, N2n);

  // ---- out = h4[cl1] * w1 ----
  gather_mul<<<G(N1n), B, 0, stream>>>(h4, cl1, w1, out, N1n, 1);
#undef ZEROF
}
